// Quantizer_52003464019991
// MI455X (gfx1250) — matrix operand order repaired and checked by reading, not compiled, not run on hardware
//
#include <hip/hip_runtime.h>
#include <hip/hip_bf16.h>
#include <stdint.h>

// ---------------------------------------------------------------------------
// VQ-VAE quantizer for MI455X (gfx1250, wave32, WMMA + TDM).
//
// Distance matmul runs as v_wmma_f32_16x16x32_bf16; bf16 codebook is staged
// through 64KB LDS in two 512-code passes moved by the Tensor Data Mover
// (tensor_load_to_lds, TENSORcnt).  Each wave owns 32 tokens (two 16x64 A
// tiles) so every B tile fetched from LDS feeds 4 WMMAs.
// x^2 is dropped from the argmin (constant per row):
//     score = cb_sq[k] - 2 * <x, cb[k]>.
// Losses / quantized output are recomputed exactly in f32 from the gather.
// ---------------------------------------------------------------------------

typedef __attribute__((ext_vector_type(16))) __bf16 v16bf;
typedef __attribute__((ext_vector_type(8)))  float  v8f;
typedef __attribute__((ext_vector_type(4)))  unsigned int u32x4;
typedef __attribute__((ext_vector_type(8)))  int  i32x8;
typedef __attribute__((ext_vector_type(4)))  int  i32x4;

#define CDIM    64
#define KCODES  1024
#define HWSZ    1024                 // 32*32
#define NTOK    65536                // 64*1024
#define NELEM   4194304              // 64*64*32*32

// workspace byte offsets
#define WS_CBB   0                   // bf16 codebook [1024][64]   = 131072 B
#define WS_CBSQ  131072              // f32 cb_sq [1024]           =   4096 B
#define WS_IDX   135168              // int32 idx [65536]          = 262144 B
#define WS_PART  397312              // f32 block partials [16384] =  65536 B

#if defined(__has_builtin)
#if __has_builtin(__builtin_amdgcn_tensor_load_to_lds)
#define VQ_HAS_TDM 1
#endif
#endif

// ---------------------------------------------------------------------------
// Kernel 1: codebook f32 -> bf16 (row major, matches WMMA B-lane chunks) and
// per-code squared norms.
// ---------------------------------------------------------------------------
__global__ void vq_prep(const float* __restrict__ cb,
                        __bf16* __restrict__ cbB,
                        float* __restrict__ cbSq) {
  int tid = blockIdx.x * blockDim.x + threadIdx.x;   // 65536 threads
  if (tid < KCODES * CDIM) cbB[tid] = (__bf16)cb[tid];
  if (tid < KCODES) {
    const float* row = cb + tid * CDIM;
    float s = 0.f;
#pragma unroll
    for (int j = 0; j < CDIM; ++j) s += row[j] * row[j];
    cbSq[tid] = s;
  }
}

// ---------------------------------------------------------------------------
// Kernel 2: argmin over codes via WMMA.
//   - 256 threads = 8 waves; wave w handles tokens [ (blk*8+w)*32 , +32 ):
//     two 16-token A tiles (tileA = hw 0..15, tileB = hw 16..31).
//   - A (16 tokens x 64 dims) in contiguous k order (same as B):
//       lane L: M = L&15, hi = L>=16; element e -> dim e + 16*hi
//       (+32 for the second K-block).
//   - B read from LDS: per lane a contiguous 16-bf16 chunk
//       cb[code][hi*16 .. hi*16+15]  (contiguous k order, matching A).
//   - C/D layout: acc[r] = row (r + 8*hi), column (lane&15).
//   - Codebook halves staged into LDS by the TDM (1-D 64KB tile descriptor),
//     issued by wave 0, completion via s_wait_tensorcnt + barrier.
// ---------------------------------------------------------------------------
__global__ __launch_bounds__(256)
void vq_argmin(const float* __restrict__ x,
               const __bf16* __restrict__ cbB,
               const float* __restrict__ cbSq,
               int* __restrict__ idxOut) {
  __shared__ __align__(128) __bf16 ldsCB[512 * CDIM];   // 64 KB: half codebook

  const int lane = threadIdx.x & 31;
  const int wave = threadIdx.x >> 5;       // 0..7
  const int hi   = lane >> 4;              // 0 / 1
  const int col  = lane & 15;

  const int tile      = blockIdx.x * 8 + wave;   // 0..2047
  const int tokenBase = tile * 32;
  const int bImg      = tokenBase >> 10;         // batch image
  const int hwBase    = tokenBase & (HWSZ - 1);  // 32-aligned, same image

  // ---- stage two A tiles (token rows: col and col+16), f32 -> bf16 ----
  const float* xb = x + (size_t)bImg * (CDIM * HWSZ) + hwBase + col;
  v16bf a0A, a1A, a0B, a1B;
#pragma unroll
  for (int e = 0; e < 16; ++e) {
    int d = e + hi * 16;
    a0A[e] = (__bf16)xb[(size_t)d * HWSZ];              // tileA dims  0..31
    a1A[e] = (__bf16)xb[(size_t)(d + 32) * HWSZ];       // tileA dims 32..63
    a0B[e] = (__bf16)xb[(size_t)d * HWSZ + 16];         // tileB dims  0..31
    a1B[e] = (__bf16)xb[(size_t)(d + 32) * HWSZ + 16];  // tileB dims 32..63
  }

  float bestA[8], bestB[8];
  int   bidxA[8], bidxB[8];
#pragma unroll
  for (int r = 0; r < 8; ++r) {
    bestA[r] = 3.4e38f; bidxA[r] = 0;
    bestB[r] = 3.4e38f; bidxB[r] = 0;
  }

  for (int pass = 0; pass < 2; ++pass) {
    __syncthreads();                           // LDS reuse protection
#ifdef VQ_HAS_TDM
    if (wave == 0) {
      // Tensor DMA descriptor: 1-D tile of 32768 2-byte elements (64 KB).
      const uint64_t g = (uint64_t)(uintptr_t)(cbB + pass * 512 * CDIM);
      const uint32_t ldsAddr = (uint32_t)(uintptr_t)(&ldsCB[0]);
      // group0: count=1 | lds_addr[63:32] | global_addr[120:64] | type=2
      u32x4 g0 = { 1u, ldsAddr, (uint32_t)g,
                   (uint32_t)(g >> 32) | 0x80000000u };
      // group1: data_size=2B; tensor_dim0=32768; tensor_dim1=1;
      //         tile_dim0=32768; tensor_dim0_stride=32768
      i32x8 g1 = { 0x00010000, (int)0x80000000u, 0x00010000,
                   (int)0x80000000u, 0, 32768, 0, 0 };
      i32x4 z4 = { 0, 0, 0, 0 };                // groups 2/3 unused (<=2-D)
      i32x8 z8 = { 0, 0, 0, 0, 0, 0, 0, 0 };
      __builtin_amdgcn_tensor_load_to_lds(g0, g1, z4, z4, z8, 0);
      __builtin_amdgcn_s_wait_tensorcnt(0);
    }
#else
    {   // fallback: cooperative 64KB copy, 4096 x uint4
      const uint4* src = (const uint4*)(cbB + pass * 512 * CDIM);
      uint4* dst = (uint4*)ldsCB;
#pragma unroll
      for (int i = 0; i < 16; ++i)
        dst[threadIdx.x + 256 * i] = src[threadIdx.x + 256 * i];
    }
#endif
    __syncthreads();

    const int codePass = pass * 512;
#pragma unroll 2
    for (int t = 0; t < 32; ++t) {
      const int code = t * 16 + col;                    // code within pass
      const __bf16* bp = ldsCB + code * CDIM + hi * 16;
      v16bf b0 = *(const v16bf*)(bp);                   // dims  0..31 chunk
      v16bf b1 = *(const v16bf*)(bp + 32);              // dims 32..63 chunk
      v8f accA = {}, accB = {};
      accA = __builtin_amdgcn_wmma_f32_16x16x32_bf16(false, a0A, false, b0,
                                                     (short)0, accA, false, false);
      accA = __builtin_amdgcn_wmma_f32_16x16x32_bf16(false, a1A, false, b1,
                                                     (short)0, accA, false, false);
      accB = __builtin_amdgcn_wmma_f32_16x16x32_bf16(false, a0B, false, b0,
                                                     (short)0, accB, false, false);
      accB = __builtin_amdgcn_wmma_f32_16x16x32_bf16(false, a1B, false, b1,
                                                     (short)0, accB, false, false);
      const int gcode = codePass + code;
      const float cs  = cbSq[gcode];
#pragma unroll
      for (int r = 0; r < 8; ++r) {
        float sA = fmaf(-2.0f, accA[r], cs);            // x^2 dropped
        float sB = fmaf(-2.0f, accB[r], cs);
        if (sA < bestA[r]) { bestA[r] = sA; bidxA[r] = gcode; }
        if (sB < bestB[r]) { bestB[r] = sB; bidxB[r] = gcode; }
      }
    }
  }

  // ---- argmin across the 16 columns of each half-wave group ----
#pragma unroll
  for (int off = 1; off < 16; off <<= 1) {
#pragma unroll
    for (int r = 0; r < 8; ++r) {
      float osA = __shfl_xor(bestA[r], off, 32);
      int   oiA = __shfl_xor(bidxA[r], off, 32);
      if (osA < bestA[r] || (osA == bestA[r] && oiA < bidxA[r])) {
        bestA[r] = osA; bidxA[r] = oiA;
      }
      float osB = __shfl_xor(bestB[r], off, 32);
      int   oiB = __shfl_xor(bidxB[r], off, 32);
      if (osB < bestB[r] || (osB == bestB[r] && oiB < bidxB[r])) {
        bestB[r] = osB; bidxB[r] = oiB;
      }
    }
  }
  if (col == 0) {
#pragma unroll
    for (int r = 0; r < 8; ++r) {
      idxOut[tokenBase + hi * 8 + r]      = bidxA[r];   // row M = r + 8*hi
      idxOut[tokenBase + 16 + hi * 8 + r] = bidxB[r];
    }
  }
}

// ---------------------------------------------------------------------------
// Kernel 3: coalesced NCHW gather of codebook rows (exact f32), squared-error
// block partials (deterministic), and idx -> float output.
// ---------------------------------------------------------------------------
__global__ __launch_bounds__(256)
void vq_gather(const float* __restrict__ x,
               const float* __restrict__ cb,
               const int* __restrict__ idxBuf,
               float* __restrict__ out,
               float* __restrict__ partial,
               float* __restrict__ idxF) {
  const int i   = blockIdx.x * 256 + threadIdx.x;   // flat NCHW index
  const int b   = i >> 16;
  const int rem = i & 65535;
  const int c   = rem >> 10;
  const int hw  = rem & 1023;
  const int tok = b * HWSZ + hw;
  const int k   = idxBuf[tok];
  const float q = cb[k * CDIM + c];
  out[i] = q;                                        // straight-through value
  const float d = x[i] - q;
  float d2 = d * d * (1.0f / (float)NELEM);

  __shared__ float red[256];
  red[threadIdx.x] = d2;
  __syncthreads();
#pragma unroll
  for (int s = 128; s > 0; s >>= 1) {
    if (threadIdx.x < s) red[threadIdx.x] += red[threadIdx.x + s];
    __syncthreads();
  }
  if (threadIdx.x == 0) partial[blockIdx.x] = red[0];
  if (c == 0) idxF[tok] = (float)k;                  // idx_emb (as f32 slot)
}

// ---------------------------------------------------------------------------
// Kernel 4: deterministic final reduction of 16384 partials -> both losses.
// ---------------------------------------------------------------------------
__global__ __launch_bounds__(256)
void vq_loss(const float* __restrict__ partial, int n,
             float* __restrict__ lossE, float* __restrict__ lossC) {
  float s = 0.f;
  for (int i = threadIdx.x; i < n; i += 256) s += partial[i];  // fixed order
  __shared__ float red[256];
  red[threadIdx.x] = s;
  __syncthreads();
#pragma unroll
  for (int st = 128; st > 0; st >>= 1) {
    if (threadIdx.x < st) red[threadIdx.x] += red[threadIdx.x + st];
    __syncthreads();
  }
  if (threadIdx.x == 0) { *lossE = red[0]; *lossC = red[0]; }
}

// ---------------------------------------------------------------------------
extern "C" void kernel_launch(void* const* d_in, const int* in_sizes, int n_in,
                              void* d_out, int out_size, void* d_ws, size_t ws_size,
                              hipStream_t stream) {
  const float* x  = (const float*)d_in[0];   // (64,64,32,32) f32
  const float* cb = (const float*)d_in[1];   // (1024,64) f32
  float* out = (float*)d_out;
  char*  ws  = (char*)d_ws;

  __bf16* cbB   = (__bf16*)(ws + WS_CBB);
  float*  cbSq  = (float*)(ws + WS_CBSQ);
  int*    idxBf = (int*)(ws + WS_IDX);
  float*  part  = (float*)(ws + WS_PART);

  float* quantOut = out;               // [4194304]
  float* lossE    = out + NELEM;       // [1]
  float* lossC    = out + NELEM + 1;   // [1]
  float* idxF     = out + NELEM + 2;   // [65536]

  vq_prep  <<<256,   256, 0, stream>>>(cb, cbB, cbSq);
  vq_argmin<<<256,   256, 0, stream>>>(x, cbB, cbSq, idxBf);
  vq_gather<<<16384, 256, 0, stream>>>(x, cb, idxBf, quantOut, part, idxF);
  vq_loss  <<<1,     256, 0, stream>>>(part, 16384, lossE, lossC);
}
